// AudioRNN_68109591380062
// MI455X (gfx1250) — matrix operand order repaired and checked by reading, not compiled, not run on hardware
//
#include <hip/hip_runtime.h>
#include <stdint.h>

typedef __attribute__((ext_vector_type(16))) _Float16 v16h;
typedef __attribute__((ext_vector_type(8)))  float    v8f;

#define I_ 13
#define H_ 32
#define C_ 4
#define T_ 25
#define B_ 65536
#define G_ 128   // 4*H

// ---- branch-free activations using native gfx1250 transcendentals ----------
__device__ __forceinline__ float fast_exp(float x) {
    // v_exp_f32 computes 2^x
    return __builtin_amdgcn_exp2f(x * 1.44269504088896340736f);
}

__device__ __forceinline__ float tanh_fast(float x) {
#if __has_builtin(__builtin_amdgcn_tanhf)
    return __builtin_amdgcn_tanhf(x);          // v_tanh_f32
#else
    const float e = fast_exp(-2.0f * __builtin_fabsf(x));
    const float r = (1.0f - e) * __builtin_amdgcn_rcpf(1.0f + e);
    return __builtin_copysignf(r, x);
#endif
}

__device__ __forceinline__ float sigf(float x) {
#if __has_builtin(__builtin_amdgcn_tanhf)
    return __builtin_fmaf(0.5f, __builtin_amdgcn_tanhf(0.5f * x), 0.5f);
#else
    return __builtin_amdgcn_rcpf(1.0f + fast_exp(-x));
#endif
}

__device__ __forceinline__ uint32_t pack2h(float a, float b) {
    union { _Float16 h[2]; uint32_t u; } t;
    t.h[0] = (_Float16)a; t.h[1] = (_Float16)b;
    return t.u;
}

// One wave32 handles 16 batch rows. Block = 256 threads = 8 waves = 128 rows.
__global__ __launch_bounds__(256) void
lstm_fc_wmma(const float* __restrict__ x,
             const float* __restrict__ W_ih,
             const float* __restrict__ W_hh,
             const float* __restrict__ b_ih,
             const float* __restrict__ b_hh,
             const float* __restrict__ W_fc,
             const float* __restrict__ b_fc,
             float* __restrict__ out)
{
    const int lane  = threadIdx.x & 31;
    const int wave  = threadIdx.x >> 5;
    const int l16   = lane & 15;
    const bool lo   = lane < 16;
    const int batch = blockIdx.x * 128 + wave * 16 + l16;

    // ---------------- Weight A-fragments (held in VGPRs for entire kernel) ---
    // A tile (f16 16x32): lane L holds row M=L%16.
    //   lanes 0-15 : v16h elts 0..7 = K 0..7,  elts 8..15 = K 16..23
    //   lanes 16-31: v16h elts 0..7 = K 8..15, elts 8..15 = K 24..31
    v16h a_hh[8], a_ih[8];
    {
        const int kb0 = lo ? 0 : 8;
        const int kb1 = lo ? 16 : 24;
        #pragma unroll
        for (int m = 0; m < 8; ++m) {
            const int g = m * 16 + l16;                 // gate row 0..127
            const float* wr = W_hh + g * H_;            // 32 contiguous floats
            const float bias = b_ih[g] + b_hh[g];
            v16h ah = {}, ai = {};
            #pragma unroll
            for (int e = 0; e < 8; ++e) {
                ah[e]     = (_Float16)wr[kb0 + e];
                ah[8 + e] = (_Float16)wr[kb1 + e];
                // W_ih padded: K<13 -> weight, K==13 -> fused bias (x K13 == 1.0),
                // K>13 -> 0. K >= 16 (kb1 range) is always 0.
                const int k = kb0 + e;
                float v = (k < I_) ? W_ih[g * I_ + k] : (k == I_ ? bias : 0.0f);
                ai[e]     = (_Float16)v;
                ai[8 + e] = (_Float16)0.0f;
            }
            a_hh[m] = ah;
            a_ih[m] = ai;
        }
    }

    // ---------------- LSTM state -------------------------------------------
    v8f  c0 = {}, c1 = {};       // cell state tiles (D layout, k 0..15 / 16..31)
    v16h bh = {};                // h B-fragment (h0 = 0)
    float hval[16];              // f32 h values, [j*8+r] <-> k = 16j + r (+8 upper)
    #pragma unroll
    for (int i = 0; i < 16; ++i) hval[i] = 0.0f;

    const float* xrow = x + (size_t)batch * (T_ * I_);
    const v8f zero8 = {};

    #pragma unroll 1
    for (int t = 0; t < T_; ++t) {
        // ---- x B-fragment: lanes<16 K=0..7, lanes>=16 K=8..12 + 1.0 at K=13; elts 8..15 0
        v16h bx = {};
        {
            const int kbx = lo ? 0 : 8;
            #pragma unroll
            for (int e = 0; e < 8; ++e) {
                const int k = kbx + e;
                bx[e] = (k < I_) ? (_Float16)xrow[t * I_ + k] : (k == I_ ? (_Float16)1.0f : (_Float16)0.0f);
            }
        }
        if (t + 1 < T_) __builtin_prefetch(xrow + (t + 1) * I_, 0, 0);

        // ---- gates^T[16g-tile, 16batch] = W_ih@[x;1] + W_hh@h  (f32 acc) ----
        v8f acc[8];
        #pragma unroll
        for (int m = 0; m < 8; ++m) {
            v8f a0 = __builtin_amdgcn_wmma_f32_16x16x32_f16(
                false, a_ih[m], false, bx, (short)0, zero8, false, false);
            acc[m] = __builtin_amdgcn_wmma_f32_16x16x32_f16(
                false, a_hh[m], false, bh, (short)0, a0, false, false);
        }

        // ---- elementwise LSTM cell (all in D layout; no cross-lane) --------
        // tiles: i = acc[0,1], f = acc[2,3], g = acc[4,5], o = acc[6,7]
        #pragma unroll
        for (int j = 0; j < 2; ++j) {
            v8f ig = acc[0 + j], fg = acc[2 + j], gg = acc[4 + j], og = acc[6 + j];
            v8f cc = j ? c1 : c0;
            #pragma unroll
            for (int r = 0; r < 8; ++r) {
                const float iv = sigf(ig[r]);
                const float fv = sigf(fg[r]);
                const float gv = tanh_fast(gg[r]);
                const float ov = sigf(og[r]);
                const float cv = fv * cc[r] + iv * gv;
                cc[r] = cv;
                hval[j * 8 + r] = ov * tanh_fast(cv);
            }
            if (j) c1 = cc; else c0 = cc;
        }

        // ---- rebuild h B-fragment: pack f16, no lane swap (D layout = B layout) -----
        uint32_t p[8];
        #pragma unroll
        for (int m = 0; m < 2; ++m)
            #pragma unroll
            for (int q = 0; q < 4; ++q)
                p[m * 4 + q] = pack2h(hval[m * 8 + 2 * q], hval[m * 8 + 2 * q + 1]);
        uint32_t s[8];
        #pragma unroll
        for (int q = 0; q < 8; ++q)
            s[q] = (uint32_t)__builtin_amdgcn_ds_swizzle((int)p[q], 0x401F); // xor 16
        union { uint32_t u[8]; v16h v; } nb;
        #pragma unroll
        for (int q = 0; q < 4; ++q) {
            nb.u[q]     = p[q];      // K 0..7  (lower) / 8..15 (upper)
            nb.u[4 + q] = p[4 + q];  // K 16..23 (lower) / 24..31 (upper)
        }
        bh = nb.v;
    }

    // ---------------- FC epilogue: out[b, c] = h . W_fc[c] + b_fc[c] --------
    float po[C_];
    #pragma unroll
    for (int c = 0; c < C_; ++c) {
        float sum = 0.0f;
        #pragma unroll
        for (int j = 0; j < 2; ++j)
            #pragma unroll
            for (int r = 0; r < 8; ++r) {
                const int k = 16 * j + r + (lo ? 0 : 8);
                sum += hval[j * 8 + r] * W_fc[c * H_ + k];
            }
        po[c] = sum;
    }
    #pragma unroll
    for (int c = 0; c < C_; ++c) {
        union { float f; uint32_t u; } t, r;
        t.f = po[c];
        r.u = (uint32_t)__builtin_amdgcn_ds_swizzle((int)t.u, 0x401F);
        po[c] += r.f;                 // combine lower/upper half partials
    }
    if (lo) {
        float4 ov;
        ov.x = po[0] + b_fc[0];
        ov.y = po[1] + b_fc[1];
        ov.z = po[2] + b_fc[2];
        ov.w = po[3] + b_fc[3];
        *(float4*)(out + (size_t)batch * C_) = ov;
    }
}

extern "C" void kernel_launch(void* const* d_in, const int* in_sizes, int n_in,
                              void* d_out, int out_size, void* d_ws, size_t ws_size,
                              hipStream_t stream) {
    (void)in_sizes; (void)n_in; (void)out_size; (void)d_ws; (void)ws_size;
    const float* x    = (const float*)d_in[0];
    const float* W_ih = (const float*)d_in[1];
    const float* W_hh = (const float*)d_in[2];
    const float* b_ih = (const float*)d_in[3];
    const float* b_hh = (const float*)d_in[4];
    const float* W_fc = (const float*)d_in[5];
    const float* b_fc = (const float*)d_in[6];
    float* out = (float*)d_out;

    dim3 grid(B_ / 128);   // 512 blocks
    dim3 block(256);       // 8 waves x 16 batch rows
    lstm_fc_wmma<<<grid, block, 0, stream>>>(x, W_ih, W_hh, b_ih, b_hh,
                                             W_fc, b_fc, out);
}
